// GNN_14164802142729
// MI455X (gfx1250) — matrix operand order repaired and checked by reading, not compiled, not run on hardware
//
#include <hip/hip_runtime.h>
#include <hip/hip_bf16.h>

// ---------------------------------------------------------------------------
// CDNA5 (gfx1250) wave32 WMMA pipeline for Conv4 + kNN-graph + 2x GATv2.
// All GEMM-shaped math uses V_WMMA_F32_16X16X32_BF16 (bf16 in, f32 acc).
// Fragments: 128-bit contiguous loads, one-deep software pipeline.
// Index math kept in 32-bit (all extents < 2^31) to avoid i64 div expansion.
// ---------------------------------------------------------------------------

typedef __attribute__((ext_vector_type(16))) __bf16 v16bf;
typedef __attribute__((ext_vector_type(8)))  __bf16 v8bf;
typedef __attribute__((ext_vector_type(8)))  float  v8f;

__device__ __forceinline__ v8f wmma_bf16(v16bf a, v16bf b, v8f c) {
  return __builtin_amdgcn_wmma_f32_16x16x32_bf16(false, a, false, b, (short)0, c,
                                                 false, false);
}
__device__ __forceinline__ v16bf cat8(v8bf lo, v8bf hi) {
  return __builtin_shufflevector(lo, hi, 0, 1, 2, 3, 4, 5, 6, 7,
                                 8, 9, 10, 11, 12, 13, 14, 15);
}

#define NIMG 256
#define BN_PARTS 1024

// ---------------------------------------------------------------------------
// conv1 im2col: x NCHW f32 [256,3,84,84] -> A1 [M1, 32] bf16
// K p-major (k = p*3 + ci), 27 real columns, zero-padded to 32.
// ---------------------------------------------------------------------------
__global__ void k_im2col1(const float* __restrict__ x, __bf16* __restrict__ A1) {
  unsigned i = blockIdx.x * blockDim.x + threadIdx.x;   // total 57.8M < 2^31
  const unsigned total = 256u * 84 * 84 * 32;
  if (i >= total) return;
  unsigned k = i & 31u;
  unsigned m = i >> 5;
  unsigned xx = m % 84u; unsigned t = m / 84u;
  unsigned yy = t % 84u; unsigned n = t / 84u;
  __bf16 v = (__bf16)0.0f;
  if (k < 27u) {
    unsigned p = k / 3u, ci = k - p * 3u;
    int dy = (int)(p / 3u) - 1, dx = (int)(p % 3u) - 1;
    int iy = (int)yy + dy, ix = (int)xx + dx;
    if (iy >= 0 && iy < 84 && ix >= 0 && ix < 84)
      v = (__bf16)x[((n * 3u + ci) * 84u + (unsigned)iy) * 84u + (unsigned)ix];
  }
  A1[i] = v;
}

// f32 [K,N] row-major -> bf16 [N,K] column-major-for-GEMM (TN layout)
__global__ void k_w_transpose_bf16(const float* __restrict__ in, __bf16* __restrict__ out,
                                   int K, int N) {
  unsigned i = blockIdx.x * blockDim.x + threadIdx.x;
  if (i >= (unsigned)(K * N)) return;
  unsigned nn = i / (unsigned)K;
  unsigned k  = i - nn * (unsigned)K;
  out[i] = (__bf16)in[k * (unsigned)N + nn];
}

// OIHW conv weight -> fragment-ready table, K ordered p-major (k = p*Cin + ci).
// Element ((c0*4 + nt)*32 + lane)*16 + e  =  W[k = c0*32 + (lane>>4)*8 + (e&7) + (e>>3)*16]
//                                            [col = nt*16 + (lane&15)], 0 past K.
__global__ void k_conv_wfrag(const float* __restrict__ w, __bf16* __restrict__ o,
                             int Cin, int K, int chunks) {
  unsigned i = blockIdx.x * blockDim.x + threadIdx.x;
  unsigned total = (unsigned)chunks * 2048u;
  if (i >= total) return;
  unsigned e    = i & 15u;
  unsigned lane = (i >> 4) & 31u;
  unsigned nt   = (i >> 9) & 3u;
  unsigned c0   = i >> 11;
  unsigned k    = c0 * 32u + ((lane >> 4) << 3) + (e & 7u) + ((e >> 3) << 4);
  unsigned col  = nt * 16u + (lane & 15u);
  __bf16 v = (__bf16)0.0f;
  if (k < (unsigned)K) {
    unsigned p = k / (unsigned)Cin, ci = k - p * (unsigned)Cin;
    v = (__bf16)w[(col * (unsigned)Cin + ci) * 9u + p];
  }
  o[i] = v;
}

// ---------------------------------------------------------------------------
// Implicit-GEMM conv -> y NHWC bf16 [M,64].
// 8 waves / block; wave = 16(M) x 64(N) strip.
// direct=1: act is an im2col matrix [M,32] (conv1), single K chunk.
// direct=0: act is NHWC [256,H,W,64], K=576, vector im2col gathers with a
//           one-deep pipelined A fragment.
// B frags come from the fragment-ready table (L1/L2 resident, b128 pairs).
// ---------------------------------------------------------------------------
__global__ void conv_gemm_wmma(const __bf16* __restrict__ act, const __bf16* __restrict__ Wf,
                               __bf16* __restrict__ y, int H, int Wd, int K, int direct) {
  int tid  = threadIdx.x;
  int wid  = tid >> 5;
  int lane = tid & 31;
  int half = lane >> 4;
  int l    = lane & 15;

  unsigned mtile = blockIdx.x * 8u + (unsigned)wid;
  unsigned m     = mtile * 16u + (unsigned)l;     // < 1.81M, fits u32

  v8f acc[4];
#pragma unroll
  for (int t = 0; t < 4; ++t) acc[t] = (v8f){};

  const __bf16* wlane = Wf + lane * 16;

  if (direct) {
    // conv1: contiguous A rows of length K (== 32, one chunk)
    const __bf16* Arow = act + (long)m * (long)K;
    v16bf a = cat8(*(const v8bf*)(Arow + half * 8),
                   *(const v8bf*)(Arow + 16 + half * 8));
#pragma unroll
    for (int nt = 0; nt < 4; ++nt) {
      v16bf b = *(const v16bf*)(wlane + nt * 512);
      acc[nt] = wmma_bf16(a, b, acc[nt]);
    }
  } else {
    // Cin = 64 NHWC gather path; 32-bit row decomposition
    unsigned hw  = (unsigned)(H * Wd);
    unsigned n   = m / hw;
    unsigned rem = m - n * hw;
    int yy = (int)(rem / (unsigned)Wd);
    int xx = (int)(rem - (unsigned)yy * (unsigned)Wd);
    long base[9];
#pragma unroll
    for (int p = 0; p < 9; ++p) {
      int dy = p / 3 - 1, dx = p % 3 - 1;
      int iy = yy + dy, ix = xx + dx;
      bool ok = (iy >= 0 && iy < H && ix >= 0 && ix < Wd);
      base[p] = ok ? ((((long)n * H + iy) * Wd + ix) * 64L) : -1;
    }
    v8bf zero = (v8bf){};
    auto loadA = [&](int k0) -> v16bf {
      int kA0 = k0 + half * 8;
      int kA1 = k0 + 16 + half * 8;
      int p0 = kA0 >> 6, ci0 = kA0 & 63;
      int p1 = kA1 >> 6, ci1 = kA1 & 63;
      v8bf lo = (base[p0] >= 0) ? *(const v8bf*)(act + base[p0] + ci0) : zero;
      v8bf hi = (base[p1] >= 0) ? *(const v8bf*)(act + base[p1] + ci1) : zero;
      return cat8(lo, hi);
    };
    v16bf a_nxt = loadA(0);
    for (int k0 = 0; k0 < K; k0 += 32) {
      v16bf a = a_nxt;
      if (k0 + 32 < K) a_nxt = loadA(k0 + 32);   // in flight during WMMAs
      const __bf16* wp = wlane + (long)(k0 >> 5) * 2048;
#pragma unroll
      for (int nt = 0; nt < 4; ++nt) {
        v16bf b = *(const v16bf*)(wp + nt * 512);
        acc[nt] = wmma_bf16(a, b, acc[nt]);
      }
    }
  }

  long mrow = (long)mtile * 16;
#pragma unroll
  for (int r = 0; r < 8; ++r) {
    long row = mrow + r + half * 8;
#pragma unroll
    for (int nt = 0; nt < 4; ++nt)
      y[row * 64 + nt * 16 + l] = (__bf16)acc[nt][r];
  }
}

// ---------------------------------------------------------------------------
// Generic TN bf16 WMMA GEMM: C[M,N] f32 = A[M,K] * B^T (B stored [N,K]).
// Requires K % 32 == 0 and M % 64 == 0. One wave = 64(M) x 16(N) strip:
// 4 M-tiles share each B fragment; fragments double-buffered one step ahead.
// ---------------------------------------------------------------------------
__global__ void gemm_bf16_wmma(const __bf16* __restrict__ A, const __bf16* __restrict__ B,
                               float* __restrict__ C, int M, int N, int K,
                               long sAm, long sBn, long sCm) {
  int gwave = (int)((blockIdx.x * blockDim.x + threadIdx.x) >> 5);
  int tilesN = N >> 4;
  int total  = (M >> 6) * tilesN;
  if (gwave >= total) return;            // uniform per wave: EXEC stays all-ones
  int bm = gwave / tilesN, tn = gwave % tilesN;
  int lane = threadIdx.x & 31;
  int half = lane >> 4;
  int l    = lane & 15;

  const __bf16* Bn = B + (long)(tn * 16 + l) * sBn;
  const __bf16* Am[4];
#pragma unroll
  for (int t = 0; t < 4; ++t)
    Am[t] = A + (long)(bm * 64 + t * 16 + l) * sAm;

  auto ldB = [&](int k0) -> v16bf {
    return cat8(*(const v8bf*)(Bn + k0 + half * 8),
                *(const v8bf*)(Bn + k0 + 16 + half * 8));
  };
  auto ldA = [&](int t, int k0) -> v16bf {
    return cat8(*(const v8bf*)(Am[t] + k0 + half * 8),
                *(const v8bf*)(Am[t] + k0 + 16 + half * 8));
  };

  v8f acc[4];
#pragma unroll
  for (int t = 0; t < 4; ++t) acc[t] = (v8f){};

  v16bf b_n = ldB(0);
  v16bf a_n[4];
#pragma unroll
  for (int t = 0; t < 4; ++t) a_n[t] = ldA(t, 0);

  for (int k0 = 0; k0 < K; k0 += 32) {
    v16bf b = b_n;
    v16bf a0 = a_n[0], a1 = a_n[1], a2 = a_n[2], a3 = a_n[3];
    if (k0 + 32 < K) {
      b_n = ldB(k0 + 32);
#pragma unroll
      for (int t = 0; t < 4; ++t) a_n[t] = ldA(t, k0 + 32);
    }
    acc[0] = wmma_bf16(a0, b, acc[0]);
    acc[1] = wmma_bf16(a1, b, acc[1]);
    acc[2] = wmma_bf16(a2, b, acc[2]);
    acc[3] = wmma_bf16(a3, b, acc[3]);
  }
#pragma unroll
  for (int t = 0; t < 4; ++t)
#pragma unroll
    for (int r = 0; r < 8; ++r)
      C[(long)(bm * 64 + t * 16 + r + half * 8) * sCm + tn * 16 + l] = acc[t][r];
}

// ---------------------------------------------------------------------------
// BatchNorm statistics, deterministic two-stage tree.
// ---------------------------------------------------------------------------
__global__ void k_bn_partial(const __bf16* __restrict__ y, float* __restrict__ part, long M) {
  int b = blockIdx.x, tid = threadIdx.x;
  long chunk = (M + BN_PARTS - 1) / BN_PARTS;
  long m0 = (long)b * chunk;
  long m1 = m0 + chunk; if (m1 > M) m1 = M;
  int c = tid & 63, sub = tid >> 6;
  float s = 0.f, s2 = 0.f;
  for (long mm = m0 + sub; mm < m1; mm += 4) {
    float v = (float)y[mm * 64 + c];
    s += v; s2 += v * v;
  }
  __shared__ float r1[256], r2[256];
  r1[tid] = s; r2[tid] = s2;
  __syncthreads();
  if (tid < 128) { r1[tid] += r1[tid + 128]; r2[tid] += r2[tid + 128]; }
  __syncthreads();
  if (tid < 64) {
    part[(long)b * 128 + tid]      = r1[tid] + r1[tid + 64];
    part[(long)b * 128 + 64 + tid] = r2[tid] + r2[tid + 64];
  }
}

__global__ void k_bn_finalize(const float* __restrict__ part, float* __restrict__ stats,
                              long M) {
  int c = blockIdx.x, tid = threadIdx.x;
  float s = 0.f, s2 = 0.f;
  for (int i = tid; i < BN_PARTS; i += 256) {
    s  += part[(long)i * 128 + c];
    s2 += part[(long)i * 128 + 64 + c];
  }
  __shared__ float r1[256], r2[256];
  r1[tid] = s; r2[tid] = s2;
  __syncthreads();
  for (int st = 128; st > 0; st >>= 1) {
    if (tid < st) { r1[tid] += r1[tid + st]; r2[tid] += r2[tid + st]; }
    __syncthreads();
  }
  if (tid == 0) {
    float mean = r1[0] / (float)M;
    float var  = r2[0] / (float)M - mean * mean;
    stats[c]      = mean;
    stats[64 + c] = rsqrtf(var + 1e-5f);
  }
}

// BN + ReLU + 2x2/2 maxpool fused (relu(max) == max(relu) by monotonicity)
__global__ void k_bn_pool(const __bf16* __restrict__ y, const float* __restrict__ stats,
                          const float* __restrict__ g, const float* __restrict__ be,
                          __bf16* __restrict__ o, int H, int W, int Ho, int Wo) {
  unsigned idx = blockIdx.x * blockDim.x + threadIdx.x;   // max 28.9M < 2^31
  unsigned total = 256u * (unsigned)(Ho * Wo) * 64u;
  if (idx >= total) return;
  unsigned c = idx & 63u; unsigned t = idx >> 6;
  unsigned xo = t % (unsigned)Wo; t /= (unsigned)Wo;
  unsigned yo = t % (unsigned)Ho; unsigned n = t / (unsigned)Ho;
  float sc = g[c] * stats[64 + c];
  float sh = be[c] - stats[c] * sc;
  float mx = 0.0f;  // ReLU floor
#pragma unroll
  for (int dy = 0; dy < 2; ++dy)
#pragma unroll
    for (int dx = 0; dx < 2; ++dx) {
      unsigned iy = 2u * yo + (unsigned)dy, ix = 2u * xo + (unsigned)dx;
      float v = (float)y[(((long)n * H + iy) * W + ix) * 64 + c] * sc + sh;
      mx = fmaxf(mx, v);
    }
  o[idx] = (__bf16)mx;
}

// pooled NHWC [256,5,5,64] -> z [256,1600] (reference NCHW flatten: c*25+y*5+x)
__global__ void k_flatten(const __bf16* __restrict__ act, float* __restrict__ z,
                          __bf16* __restrict__ zb) {
  unsigned i = blockIdx.x * blockDim.x + threadIdx.x;
  if (i >= 256u * 1600u) return;
  unsigned n = i / 1600u;
  unsigned r = i - n * 1600u;
  unsigned c = r / 25u, p = r - c * 25u, yy = p / 5u, xx = p - yy * 5u;
  float v = (float)act[((n * 5u + yy) * 5u + xx) * 64u + c];
  z[i]  = v;
  zb[i] = (__bf16)v;
}

// row-normalize z (cosine prep): zn = z * rsqrt(sum z^2 + 1e-12), bf16 out
__global__ void k_make_zn(const float* __restrict__ z, __bf16* __restrict__ zn) {
  int n = blockIdx.x, tid = threadIdx.x;
  __shared__ float red[256];
  __shared__ float rs;
  float s = 0.f;
  for (int c = tid; c < 1600; c += 256) { float v = z[n * 1600 + c]; s += v * v; }
  red[tid] = s;
  __syncthreads();
  for (int st = 128; st > 0; st >>= 1) {
    if (tid < st) red[tid] += red[tid + st];
    __syncthreads();
  }
  if (tid == 0) rs = rsqrtf(red[0] + 1e-12f);
  __syncthreads();
  for (int c = tid; c < 1600; c += 256)
    zn[n * 1600 + c] = (__bf16)(z[n * 1600 + c] * rs);
}

// top-10 per row of sim (self masked), one wave per node, shuffle argmax
__global__ void k_topk(const float* __restrict__ sim, int* __restrict__ nbr) {
  int n = blockIdx.x, lane = threadIdx.x;
  float v[8];
#pragma unroll
  for (int i = 0; i < 8; ++i) {
    int col = lane * 8 + i;
    float s = sim[n * 256 + col];
    v[i] = (col == n) ? -3.0e38f : s;
  }
  if (lane == 0) nbr[n * 11] = n;
  for (int t = 0; t < 10; ++t) {
    float bm = -3.0e38f; int bi = 0x7fffffff;
#pragma unroll
    for (int i = 0; i < 8; ++i)
      if (v[i] > bm) { bm = v[i]; bi = lane * 8 + i; }
    for (int off = 16; off; off >>= 1) {
      float om = __shfl_xor(bm, off);
      int   oi = __shfl_xor(bi, off);
      if (om > bm || (om == bm && oi < bi)) { bm = om; bi = oi; }
    }
    if (lane == 0) nbr[n * 11 + 1 + t] = bi;
    int li = bi - lane * 8;
    if (li >= 0 && li < 8) v[li] = -3.0e38f;
  }
}

// ---------------------------------------------------------------------------
// GATv2 attention: one block (256 thr) per node. xl = source/messages, xr = target.
// concatRelu=1: out[n, h*C+c] = relu(sum_k a*xl + bias);  =0: head-mean + bias.
// ---------------------------------------------------------------------------
__global__ void k_gatv2(const float* __restrict__ xl, const float* __restrict__ xr,
                        const float* __restrict__ att, const int* __restrict__ nbr,
                        const float* __restrict__ bias, float* __restrict__ outF,
                        __bf16* __restrict__ outB, int Hh, int C, int concatRelu) {
  int n = blockIdx.x, tid = threadIdx.x;
  __shared__ float red[256];
  __shared__ float alpha[11];
  float acc[8];
#pragma unroll
  for (int i = 0; i < 8; ++i) acc[i] = 0.f;
  int idxs[11];
#pragma unroll
  for (int k = 0; k < 11; ++k) idxs[k] = nbr[n * 11 + k];

  for (int h = 0; h < Hh; ++h) {
    const float* xrn = xr + (n * Hh + h) * C;
    float lg[11];
    for (int k = 0; k < 11; ++k) {
      const float* xlj = xl + (idxs[k] * Hh + h) * C;
      float p = 0.f;
      for (int c = tid; c < C; c += 256) {
        float e = xrn[c] + xlj[c];
        e = (e > 0.f) ? e : 0.2f * e;     // leaky_relu(0.2)
        p += e * att[h * C + c];
      }
      red[tid] = p;
      __syncthreads();
      for (int st = 128; st; st >>= 1) {
        if (tid < st) red[tid] += red[tid + st];
        __syncthreads();
      }
      lg[k] = red[0];
      __syncthreads();
    }
    if (tid == 0) {
      float mx = lg[0];
      for (int k = 1; k < 11; ++k) mx = fmaxf(mx, lg[k]);
      float s = 0.f;
      for (int k = 0; k < 11; ++k) { float a = __expf(lg[k] - mx); alpha[k] = a; s += a; }
      float inv = 1.f / s;
      for (int k = 0; k < 11; ++k) alpha[k] *= inv;
    }
    __syncthreads();
    int ai = 0;
    for (int c = tid; c < C; c += 256, ++ai) {
      float o = 0.f;
#pragma unroll
      for (int k = 0; k < 11; ++k)
        o += alpha[k] * xl[(idxs[k] * Hh + h) * C + c];
      if (concatRelu) {
        float vv = o + bias[h * C + c];
        vv = vv > 0.f ? vv : 0.f;
        outF[(n * Hh + h) * C + c] = vv;
        if (outB) outB[(n * Hh + h) * C + c] = (__bf16)vv;
      } else {
        acc[ai] += o;
      }
    }
    __syncthreads();
  }
  if (!concatRelu) {
    float invH = 1.f / (float)Hh;
    int ai = 0;
    for (int c = tid; c < C; c += 256, ++ai)
      outF[n * C + c] = acc[ai] * invH + bias[c];
  }
}

// ---------------------------------------------------------------------------
// Host orchestration
// ---------------------------------------------------------------------------
extern "C" void kernel_launch(void* const* d_in, const int* in_sizes, int n_in,
                              void* d_out, int out_size, void* d_ws, size_t ws_size,
                              hipStream_t stream) {
  (void)in_sizes; (void)n_in; (void)out_size; (void)ws_size;
  const float* x   = (const float*)d_in[0];
  const float* cw[4] = {(const float*)d_in[1], (const float*)d_in[5],
                        (const float*)d_in[9], (const float*)d_in[13]};
  const float* bg[4] = {(const float*)d_in[3], (const float*)d_in[7],
                        (const float*)d_in[11], (const float*)d_in[15]};
  const float* bb[4] = {(const float*)d_in[4], (const float*)d_in[8],
                        (const float*)d_in[12], (const float*)d_in[16]};
  const float* wl1 = (const float*)d_in[17];
  const float* wr1 = (const float*)d_in[18];
  const float* att1  = (const float*)d_in[19];
  const float* bias1 = (const float*)d_in[20];
  const float* wl2 = (const float*)d_in[21];
  const float* wr2 = (const float*)d_in[22];
  const float* att2  = (const float*)d_in[23];
  const float* bias2 = (const float*)d_in[24];
  float* out = (float*)d_out;

  // ---- workspace layout ----
  // A1 (conv1 im2col, 115.6 MB) aliases actA+actB exactly: A1 is dead before
  // pool1 first writes actA (stream order guarantees safety).
  char* base = (char*)d_ws;
  size_t off = 0;
  auto alloc = [&](size_t bytes) -> char* {
    char* p = base + off;
    off = (off + bytes + 255) & ~(size_t)255;
    return p;
  };
  __bf16* Y    = (__bf16*)alloc(256L * 84 * 84 * 64 * 2);   // pre-BN conv out (max size)
  __bf16* actA = (__bf16*)alloc(256L * 42 * 42 * 64 * 2);   // ping  (sizes are 256B-
  __bf16* actB = (__bf16*)alloc(256L * 42 * 42 * 64 * 2);   // pong   multiples: adjacent)
  __bf16* A1   = actA;                                      // 256*84*84*32*2 == both
  __bf16* wB[4];
  wB[0] = (__bf16*)alloc(2048L * 2);                        // 1 chunk (K=27 padded)
  for (int i = 1; i < 4; ++i) wB[i] = (__bf16*)alloc(18L * 2048 * 2);  // 18 chunks
  float*  part  = (float*)alloc((long)BN_PARTS * 128 * 4);
  float*  stats = (float*)alloc(128 * 4);
  __bf16* zbf   = (__bf16*)alloc(256L * 1600 * 2);
  __bf16* znbf  = (__bf16*)alloc(256L * 1600 * 2);
  float*  sim   = (float*)alloc(256L * 256 * 4);
  int*    nbr   = (int*)alloc(256L * 11 * 4);
  __bf16* wl1T  = (__bf16*)alloc(400L * 1600 * 2);   // [N=400, K=1600]
  __bf16* wr1T  = (__bf16*)alloc(400L * 1600 * 2);
  __bf16* wl2T  = (__bf16*)alloc(6400L * 400 * 2);   // [N=6400, K=400]
  __bf16* wr2T  = (__bf16*)alloc(6400L * 400 * 2);
  float*  xl1   = (float*)alloc(256L * 400 * 4);
  float*  xr1   = (float*)alloc(256L * 400 * 4);
  float*  h1f   = (float*)alloc(256L * 400 * 4);
  __bf16* h1b   = (__bf16*)alloc(256L * 400 * 2);
  float*  xl2   = (float*)alloc(256L * 6400 * 4);
  float*  xr2   = (float*)alloc(256L * 6400 * 4);

  auto cdiv = [](long a, long b) { return (int)((a + b - 1) / b); };

  // ---- weight conversion (independent of activations) ----
  k_conv_wfrag<<<cdiv(2048, 256), 256, 0, stream>>>(cw[0], wB[0], 3, 27, 1);
  for (int i = 1; i < 4; ++i)
    k_conv_wfrag<<<cdiv(18L * 2048, 256), 256, 0, stream>>>(cw[i], wB[i], 64, 576, 18);
  k_w_transpose_bf16<<<cdiv(1600L * 400, 256), 256, 0, stream>>>(wl1, wl1T, 1600, 400);
  k_w_transpose_bf16<<<cdiv(1600L * 400, 256), 256, 0, stream>>>(wr1, wr1T, 1600, 400);
  k_w_transpose_bf16<<<cdiv(400L * 6400, 256), 256, 0, stream>>>(wl2, wl2T, 400, 6400);
  k_w_transpose_bf16<<<cdiv(400L * 6400, 256), 256, 0, stream>>>(wr2, wr2T, 400, 6400);

  // ---- Conv4 backbone: 84 -> 42 -> 21 -> 10 -> 5 (conv bias absorbed by BN) ----
  k_im2col1<<<cdiv(256L * 84 * 84 * 32, 256), 256, 0, stream>>>(x, A1);

  struct Layer { int H, Ho, K, direct; __bf16 *in, *outp; };
  Layer L[4] = {
    {84, 42, 32,  1, A1,   actA},   // conv1 consumes A1, then pool1 overwrites it
    {42, 21, 576, 0, actA, actB},
    {21, 10, 576, 0, actB, actA},
    {10, 5,  576, 0, actA, actB},
  };
  for (int l = 0; l < 4; ++l) {
    long M = 256L * L[l].H * L[l].H;                 // divisible by 128 for all layers
    conv_gemm_wmma<<<(int)(M / 128), 256, 0, stream>>>(L[l].in, wB[l], Y,
                                                       L[l].H, L[l].H, L[l].K, L[l].direct);
    k_bn_partial<<<BN_PARTS, 256, 0, stream>>>(Y, part, M);
    k_bn_finalize<<<64, 256, 0, stream>>>(part, stats, M);
    long tot = 256L * L[l].Ho * L[l].Ho * 64;
    k_bn_pool<<<cdiv(tot, 256), 256, 0, stream>>>(Y, stats, bg[l], bb[l], L[l].outp,
                                                  L[l].H, L[l].H, L[l].Ho, L[l].Ho);
  }

  // ---- flatten -> z_cnn (first output) + bf16 copy ----
  k_flatten<<<cdiv(256L * 1600, 256), 256, 0, stream>>>(actB, out, zbf);

  // ---- cosine-kNN graph ----
  k_make_zn<<<256, 256, 0, stream>>>(out, znbf);
  auto gemm = [&](const __bf16* A, const __bf16* B, float* C, int M, int N, int K,
                  long sAm, long sBn, long sCm) {
    int waves = (M / 64) * (N / 16);
    gemm_bf16_wmma<<<(waves + 7) / 8, 256, 0, stream>>>(A, B, C, M, N, K, sAm, sBn, sCm);
  };
  // sim = zn * zn^T  (zn rows are already the [N,K] layout for B)
  gemm(znbf, znbf, sim, 256, 256, 1600, 1600, 1600, 256);
  k_topk<<<256, 32, 0, stream>>>(sim, nbr);

  // ---- GATv2 layer 1: EMB(1600) -> HID(400), 4 heads x 100, concat + ReLU ----
  gemm(zbf, wl1T, xl1, 256, 400, 1600, 1600, 1600, 400);
  gemm(zbf, wr1T, xr1, 256, 400, 1600, 1600, 1600, 400);
  k_gatv2<<<256, 256, 0, stream>>>(xl1, xr1, att1, nbr, bias1, h1f, h1b, 4, 100, 1);

  // ---- GATv2 layer 2: HID(400) -> EMB(1600), 4 heads x 1600, head-mean ----
  gemm(h1b, wl2T, xl2, 256, 6400, 400, 400, 400, 6400);
  gemm(h1b, wr2T, xr2, 256, 6400, 400, 400, 400, 6400);
  k_gatv2<<<256, 256, 0, stream>>>(xl2, xr2, att2, nbr, bias2,
                                   out + 256L * 1600, nullptr, 4, 1600, 0);
}
